// GPTQMarlinMoE_18287970746808
// MI455X (gfx1250) — matrix operand order repaired and checked by reading, not compiled, not run on hardware
//
#include <hip/hip_runtime.h>

#define E_      8
#define K_      2048
#define N_      2048
#define TWO_N   4096
#define G1_     16
#define G2_     16
#define T_      2048
#define TOPK_   2
#define MAXTILES 136            // T*TOPK/32 + E (worst-case padding)
#define MAXSLOTS (MAXTILES * 32)

typedef __attribute__((ext_vector_type(16))) __bf16 v16bf;
typedef __attribute__((ext_vector_type(2)))  __bf16 v2bf;
typedef __attribute__((ext_vector_type(2)))  float  v2f;
typedef __attribute__((ext_vector_type(8)))  float  v8f;

struct alignas(16) U4 { unsigned x, y, z, w; };
struct alignas(8)  U2 { unsigned x, y; };
struct alignas(16) F4 { float x, y, z, w; };

union Frag16 { v16bf v; U4 q[2]; };

__device__ __forceinline__ unsigned short f2bf(float f) {
  unsigned int u = __builtin_bit_cast(unsigned int, f);
  u += 0x7FFFu + ((u >> 16) & 1u);          // round-to-nearest-even
  return (unsigned short)(u >> 16);
}

// pack two floats into one bf16x2 dword; convertvector lets the backend pick
// the native (packed) f32->bf16 converter instead of manual bit twiddling
__device__ __forceinline__ unsigned pk2bf(float a, float b) {
  v2f f = {a, b};
  v2bf r = __builtin_convertvector(f, v2bf);
  return __builtin_bit_cast(unsigned, r);
}

__device__ __forceinline__ v2bf bf16x2_splat(float f) {
  __bf16 b = __builtin_bit_cast(__bf16, f2bf(f));
  v2bf r = {b, b};
  return r;
}

// Marlin-style int4 dequant: 0x4300|q is exactly bf16(128+q), so
// (q-(z+1))*s == fma_bf16(0x4300|q, s, -(128+z+1)*s). One packed FMA per pair.
__device__ __forceinline__ unsigned dq2(unsigned word, int sh, v2bf spk, v2bf dpk) {
  unsigned lo = (word >> sh) & 0xFu;
  unsigned hi = (word >> (sh + 4)) & 0xFu;
  unsigned t  = 0x43004300u | lo | (hi << 16);
  v2bf tv = __builtin_bit_cast(v2bf, t);
  v2bf r  = __builtin_elementwise_fma(tv, spk, dpk);   // v_pk_fma_bf16
  return __builtin_bit_cast(unsigned, r);
}

__device__ __forceinline__ v8f wmma_bf16(v16bf a, v16bf b, v8f c) {
  return __builtin_amdgcn_wmma_f32_16x16x32_bf16(
      /*neg_a=*/false, a, /*neg_b=*/false, b,
      /*c_mod=*/(short)0, c, /*reuse_a=*/false, /*reuse_b=*/false);
}

// ---------------------------------------------------------------- routing ---
__global__ __launch_bounds__(256) void route_kernel(
    const int* __restrict__ topk_ids, const float* __restrict__ topk_w,
    int* __restrict__ tile_expert, int* __restrict__ slot_token,
    float* __restrict__ slot_w) {
  __shared__ int cnt[E_], cnt2[E_], base[E_];
  const int tid = threadIdx.x;
  if (tid < E_) { cnt[tid] = 0; cnt2[tid] = 0; }
  __syncthreads();
  for (int i = tid; i < T_ * TOPK_; i += blockDim.x)
    atomicAdd(&cnt[topk_ids[i]], 1);
  __syncthreads();
  if (tid == 0) {
    int b = 0;
    for (int e = 0; e < E_; ++e) { base[e] = b; b += ((cnt[e] + 31) >> 5) << 5; }
  }
  __syncthreads();
  for (int s = tid; s < MAXSLOTS; s += blockDim.x) { slot_token[s] = -1; slot_w[s] = 0.f; }
  for (int t = tid; t < MAXTILES; t += blockDim.x) {
    int te = -1;
    for (int e = 0; e < E_; ++e) {
      int t0 = base[e] >> 5;
      int nt = (cnt[e] + 31) >> 5;
      if (t >= t0 && t < t0 + nt) te = e;
    }
    tile_expert[t] = te;
  }
  __syncthreads();
  for (int i = tid; i < T_ * TOPK_; i += blockDim.x) {
    int e = topk_ids[i];
    int p = atomicAdd(&cnt2[e], 1);
    int s = base[e] + p;
    slot_token[s] = i / TOPK_;
    slot_w[s] = topk_w[i];
  }
}

__global__ __launch_bounds__(256) void zero_kernel(float* __restrict__ out, int n) {
  int i = blockIdx.x * blockDim.x + threadIdx.x;
  if (i < n) out[i] = 0.f;
}

// ------------------------------------------------------- GEMM1 (x@W1, SiLU) -
__global__ __launch_bounds__(256) void gemm1_kernel(
    const float* __restrict__ x, const int* __restrict__ qw,
    const int* __restrict__ qz, const float* __restrict__ sc,
    const int* __restrict__ tile_expert, const int* __restrict__ slot_token,
    unsigned short* __restrict__ a_buf) {
  __shared__ unsigned short sA[32][72];    // x tile (row x 64k), bf16, 16B-aligned rows
  __shared__ unsigned short sB[128][72];   // W tile transposed (col x 64k), bf16
  __shared__ float          sH[32][128];   // h tile (gate|up)
  __shared__ int            sTok[32];

  const int ct = blockIdx.x;               // a-col tile: cols [ct*64, ct*64+64)
  const int mt = blockIdx.y;               // slot tile (32 rows)
  const int e  = tile_expert[mt];
  if (e < 0) return;

  const int tid = threadIdx.x;
  if (tid < 32) sTok[tid] = slot_token[mt * 32 + tid];
  __syncthreads();

  const int w = tid >> 5, lane = tid & 31, lhi = lane >> 4, lr = lane & 15;
  const int bcol = w * 16 + lr;
  const int colA = ct * 64;

  // ---- hoisted A-gather state: each thread owns a float4 in rows r0 and r1
  const int r0 = tid >> 4, r1 = r0 + 16;
  const int kq = (tid & 15) * 4;
  const int t0 = sTok[r0], t1 = sTok[r1];
  const float m0 = (t0 >= 0) ? 1.f : 0.f;
  const float m1 = (t1 >= 0) ? 1.f : 0.f;
  const float* xp0 = x + (size_t)(t0 >= 0 ? t0 : 0) * K_ + kq;
  const float* xp1 = x + (size_t)(t1 >= 0 ? t1 : 0) * K_ + kq;

  // ---- hoisted B-dequant state: thread owns column c, k-half hv (4 words)
  const int c = tid & 127, hv = tid >> 7;
  const int hcol = (c < 64) ? (colA + c) : (N_ + colA + (c - 64));  // col in [0,2N)
  const float* scp = sc + (size_t)e * G1_ * TWO_N + hcol;
  const int*   qzp = qz + e * G1_ * (TWO_N / 8) + (hcol >> 3);
  const int    zsh = (hcol & 7) * 4;
  const int*   qwp = qw + (size_t)e * (K_ / 8) * TWO_N + (size_t)hv * 4 * TWO_N + hcol;

  v8f acc0 = {}; v8f acc1 = {};

#pragma unroll 1
  for (int g = 0; g < G1_; ++g) {          // one GPTQ group = 128 k
    const float s = scp[(size_t)g * TWO_N];
    const unsigned zw = (unsigned)qzp[g * (TWO_N / 8)];
    const float z1 = (float)(((zw >> zsh) & 0xF) + 1);
    // use the bf16-rounded scale for the additive term so fma is consistent
    const float srnd = __builtin_bit_cast(float, (unsigned)f2bf(s) << 16);
    const v2bf spk = bf16x2_splat(s);
    const v2bf dpk = bf16x2_splat(-srnd * (128.f + z1));
#pragma unroll
    for (int hf = 0; hf < 2; ++hf) {       // two 64-k steps per group
      const int k0 = g * 128 + hf * 64;
      // stage A: 2x float4 coalesced, masked, packed to bf16 (uint2 -> ds b64)
      F4 xa = *(const F4*)(xp0 + k0);
      F4 xb = *(const F4*)(xp1 + k0);
      U2 pa; pa.x = pk2bf(xa.x * m0, xa.y * m0); pa.y = pk2bf(xa.z * m0, xa.w * m0);
      U2 pb; pb.x = pk2bf(xb.x * m1, xb.y * m1); pb.y = pk2bf(xb.z * m1, xb.w * m1);
      *(U2*)&sA[r0][kq] = pa;
      *(U2*)&sA[r1][kq] = pb;

      // stage B: dequant 4 int4 words -> 4x ds_store_b128 (v_pk_fma_bf16 path)
      const int* qrow = qwp + (size_t)(k0 >> 3) * TWO_N;
#pragma unroll
      for (int j = 0; j < 4; ++j) {
        const unsigned word = (unsigned)qrow[(size_t)j * TWO_N];
        U4 o;
        o.x = dq2(word, 0,  spk, dpk);
        o.y = dq2(word, 8,  spk, dpk);
        o.z = dq2(word, 16, spk, dpk);
        o.w = dq2(word, 24, spk, dpk);
        *(U4*)&sB[c][hv * 32 + j * 8] = o;
      }
      if (k0 + 64 < K_)
        __builtin_prefetch(qrow + (size_t)8 * TWO_N, 0, 0);
      __syncthreads();

      // compute: 2 k-substeps x 2 row tiles = 4 WMMAs; frags are b128 LDS loads
#pragma unroll
      for (int ks = 0; ks < 2; ++ks) {
        const int kb = ks * 32;
        Frag16 fa0, fa1, fb;
        fa0.q[0] = *(const U4*)&sA[lr][kb + lhi * 8];
        fa0.q[1] = *(const U4*)&sA[lr][kb + 16 + lhi * 8];
        fa1.q[0] = *(const U4*)&sA[16 + lr][kb + lhi * 8];
        fa1.q[1] = *(const U4*)&sA[16 + lr][kb + 16 + lhi * 8];
        fb.q[0]  = *(const U4*)&sB[bcol][kb + lhi * 8];
        fb.q[1]  = *(const U4*)&sB[bcol][kb + 16 + lhi * 8];
        acc0 = wmma_bf16(fa0.v, fb.v, acc0);
        acc1 = wmma_bf16(fa1.v, fb.v, acc1);
      }
      __syncthreads();
    }
  }

  // C/D layout: VGPR r -> row r (lanes 0-15) / row r+8 (lanes 16-31)
#pragma unroll
  for (int r = 0; r < 8; ++r) {
    int row = (lhi << 3) + r;
    sH[row][w * 16 + lr]      = acc0[r];
    sH[16 + row][w * 16 + lr] = acc1[r];
  }
  __syncthreads();

  // fused SiLU(gate)*up, pair-packed b32 stores to the bf16 activation buffer
#pragma unroll
  for (int i = 0; i < 4; ++i) {
    int idx = tid + i * 256;                  // 1024 pairs over 32x64
    int r = idx >> 5, cc = (idx & 31) * 2;
    float g0 = sH[r][cc],     u0 = sH[r][64 + cc];
    float g1 = sH[r][cc + 1], u1 = sH[r][64 + cc + 1];
    float a0 = (g0 / (1.f + __expf(-g0))) * u0;
    float a1 = (g1 / (1.f + __expf(-g1))) * u1;
    *(unsigned*)&a_buf[((size_t)(mt * 32 + r)) * N_ + colA + cc] = pk2bf(a0, a1);
  }
}

// --------------------------------------------- GEMM2 (a@W2, routed scatter) -
__global__ __launch_bounds__(256) void gemm2_kernel(
    const unsigned short* __restrict__ a_buf, const int* __restrict__ qw,
    const int* __restrict__ qz, const float* __restrict__ sc,
    const int* __restrict__ tile_expert, const int* __restrict__ slot_token,
    const float* __restrict__ slot_w, float* __restrict__ out) {
  __shared__ unsigned short sA[32][72];
  __shared__ unsigned short sB[128][72];
  __shared__ float          sO[32][128];
  __shared__ int            sTok[32];
  __shared__ float          sWt[32];

  const int ct = blockIdx.x;               // out-col tile: [ct*128, ct*128+128)
  const int mt = blockIdx.y;
  const int e  = tile_expert[mt];
  if (e < 0) return;

  const int tid = threadIdx.x;
  if (tid < 32) { sTok[tid] = slot_token[mt * 32 + tid]; sWt[tid] = slot_w[mt * 32 + tid]; }
  __syncthreads();

  const int w = tid >> 5, lane = tid & 31, lhi = lane >> 4, lr = lane & 15;
  const int bcol = w * 16 + lr;

  // A staging (bf16 already): 2x uint2 per thread per step
  const int r0 = tid >> 4, r1 = r0 + 16;
  const int kq = (tid & 15) * 4;
  const unsigned short* ap0 = a_buf + ((size_t)(mt * 32 + r0)) * N_ + kq;
  const unsigned short* ap1 = a_buf + ((size_t)(mt * 32 + r1)) * N_ + kq;

  // B dequant state
  const int c = tid & 127, hv = tid >> 7;
  const int kcol = ct * 128 + c;
  const float* scp = sc + (size_t)e * G2_ * K_ + kcol;
  const int*   qzp = qz + e * G2_ * (K_ / 8) + (kcol >> 3);
  const int    zsh = (kcol & 7) * 4;
  const int*   qwp = qw + (size_t)e * (N_ / 8) * K_ + (size_t)hv * 4 * K_ + kcol;

  v8f acc0 = {}; v8f acc1 = {};

#pragma unroll 1
  for (int g = 0; g < G2_; ++g) {          // one GPTQ group = 128 n
    const float s = scp[(size_t)g * K_];
    const unsigned zw = (unsigned)qzp[g * (K_ / 8)];
    const float z1 = (float)(((zw >> zsh) & 0xF) + 1);
    const float srnd = __builtin_bit_cast(float, (unsigned)f2bf(s) << 16);
    const v2bf spk = bf16x2_splat(s);
    const v2bf dpk = bf16x2_splat(-srnd * (128.f + z1));
#pragma unroll
    for (int hf = 0; hf < 2; ++hf) {
      const int n0 = g * 128 + hf * 64;
      *(U2*)&sA[r0][kq] = *(const U2*)(ap0 + n0);
      *(U2*)&sA[r1][kq] = *(const U2*)(ap1 + n0);

      const int* qrow = qwp + (size_t)(n0 >> 3) * K_;
#pragma unroll
      for (int j = 0; j < 4; ++j) {
        const unsigned word = (unsigned)qrow[(size_t)j * K_];
        U4 o;
        o.x = dq2(word, 0,  spk, dpk);
        o.y = dq2(word, 8,  spk, dpk);
        o.z = dq2(word, 16, spk, dpk);
        o.w = dq2(word, 24, spk, dpk);
        *(U4*)&sB[c][hv * 32 + j * 8] = o;
      }
      if (n0 + 64 < N_)
        __builtin_prefetch(qrow + (size_t)8 * K_, 0, 0);
      __syncthreads();

#pragma unroll
      for (int ks = 0; ks < 2; ++ks) {
        const int kb = ks * 32;
        Frag16 fa0, fa1, fb;
        fa0.q[0] = *(const U4*)&sA[lr][kb + lhi * 8];
        fa0.q[1] = *(const U4*)&sA[lr][kb + 16 + lhi * 8];
        fa1.q[0] = *(const U4*)&sA[16 + lr][kb + lhi * 8];
        fa1.q[1] = *(const U4*)&sA[16 + lr][kb + 16 + lhi * 8];
        fb.q[0]  = *(const U4*)&sB[bcol][kb + lhi * 8];
        fb.q[1]  = *(const U4*)&sB[bcol][kb + 16 + lhi * 8];
        acc0 = wmma_bf16(fa0.v, fb.v, acc0);
        acc1 = wmma_bf16(fa1.v, fb.v, acc1);
      }
      __syncthreads();
    }
  }

#pragma unroll
  for (int r = 0; r < 8; ++r) {
    int row = (lhi << 3) + r;
    sO[row][w * 16 + lr]      = acc0[r];
    sO[16 + row][w * 16 + lr] = acc1[r];
  }
  __syncthreads();

#pragma unroll
  for (int i = 0; i < 16; ++i) {
    int idx = tid + i * 256;
    int r = idx >> 7, cc = idx & 127;
    int tok = sTok[r];
    if (tok >= 0)
      atomicAdd(&out[(size_t)tok * K_ + ct * 128 + cc], sWt[r] * sO[r][cc]);
  }
}

// ------------------------------------------------------------------ launch --
extern "C" void kernel_launch(void* const* d_in, const int* in_sizes, int n_in,
                              void* d_out, int out_size, void* d_ws, size_t ws_size,
                              hipStream_t stream) {
  const float* x      = (const float*)d_in[0];
  const int*   w1_qw  = (const int*)d_in[1];
  const int*   w1_qz  = (const int*)d_in[2];
  const float* w1_sc  = (const float*)d_in[3];
  const int*   w2_qw  = (const int*)d_in[6];
  const int*   w2_qz  = (const int*)d_in[7];
  const float* w2_sc  = (const float*)d_in[8];
  const float* tkw    = (const float*)d_in[11];
  const int*   tki    = (const int*)d_in[12];
  float* out = (float*)d_out;

  // workspace layout
  char* ws = (char*)d_ws;
  int*   tile_expert = (int*)ws;                         // 256 ints (136 used)
  int*   slot_token  = tile_expert + 256;                // MAXSLOTS ints
  float* slot_w      = (float*)(slot_token + MAXSLOTS);  // MAXSLOTS floats
  unsigned short* a_buf = (unsigned short*)(slot_w + MAXSLOTS); // MAXSLOTS*N bf16

  route_kernel<<<1, 256, 0, stream>>>(tki, tkw, tile_expert, slot_token, slot_w);
  zero_kernel<<<(T_ * K_ + 255) / 256, 256, 0, stream>>>(out, T_ * K_);
  gemm1_kernel<<<dim3(TWO_N / 128, MAXTILES), 256, 0, stream>>>(
      x, w1_qw, w1_qz, w1_sc, tile_expert, slot_token, a_buf);
  gemm2_kernel<<<dim3(K_ / 128, MAXTILES), 256, 0, stream>>>(
      a_buf, w2_qw, w2_qz, w2_sc, tile_expert, slot_token, slot_w, out);
}
